// FlashLinearAttentionNet_17085379904256
// MI455X (gfx1250) — matrix operand order repaired and checked by reading, not compiled, not run on hardware
//
#include <hip/hip_runtime.h>

constexpr int Bn = 16;
constexpr int Tn = 16384;
constexpr int Dn = 128;
constexpr int Hn = 256;
constexpr int WGPB  = 32;                 // workgroups per batch (phases 1 & 2)
constexpr int TILES = (Tn / WGPB) / 128;  // 4 tiles of 128 rows per WG
constexpr int LS    = 136;                // padded LDS row stride (bf16 elems); 68 dwords ≡ 4 (mod 64)

typedef __bf16 bf16_t;
typedef __attribute__((ext_vector_type(16))) __bf16 v16bf;
typedef __attribute__((ext_vector_type(8)))  __bf16 bf16x8;
typedef __attribute__((ext_vector_type(8)))  float  v8f;

__device__ inline v8f wmma_bf16(v16bf a, v16bf b, v8f c) {
  // D = A(16x32 bf16) * B(32x16 bf16) + C(16x16 f32)
  return __builtin_amdgcn_wmma_f32_16x16x32_bf16(false, a, false, b, (short)0, c, false, false);
}

// A-operand (16x32, 16-bit): lane = row (mod 16); lo-half lanes K {0..7,16..23}, hi-half {8..15,24..31}
__device__ inline v16bf load_a(const bf16_t* base, int row, int kc, int hi) {
  const bf16_t* p = base + row * LS + kc * 32 + hi * 8;
  bf16x8 lo = *(const bf16x8*)(p);
  bf16x8 h8 = *(const bf16x8*)(p + 16);
  return __builtin_shufflevector(lo, h8, 0,1,2,3,4,5,6,7,8,9,10,11,12,13,14,15);
}

// B-operand (32x16, 16-bit): lane = column (mod 16); lo-half lanes K {0..7,16..23}, hi-half {8..15,24..31}
__device__ inline v16bf load_b(const bf16_t* base, int row, int kc, int hi) {
  const bf16_t* p = base + row * LS + kc * 32 + hi * 8;
  bf16x8 lo = *(const bf16x8*)(p);
  bf16x8 h8 = *(const bf16x8*)(p + 16);
  return __builtin_shufflevector(lo, h8, 0,1,2,3,4,5,6,7,8,9,10,11,12,13,14,15);
}

__device__ inline v8f splat8(float f) {
  v8f r;
#pragma unroll
  for (int i = 0; i < 8; ++i) r[i] = f;
  return r;
}

__device__ inline void store8(bf16_t* dst, v8f c) {  // packed 16B store (D-layout natural transpose)
  bf16x8 o;
#pragma unroll
  for (int i = 0; i < 8; ++i) o[i] = (bf16_t)c[i];
  *(bf16x8*)dst = o;
}

__device__ inline void ld_weight(bf16_t* dst, const bf16_t* src, int tid) {
  for (int i = tid; i < Dn * 16; i += 256) {
    int r = i >> 4, c = (i & 15) * 8;
    *(bf16x8*)(dst + r * LS + c) = *(const bf16x8*)(src + r * Dn + c);
  }
}

__device__ inline void ld_f32_tile(bf16_t* dst, const float* src, int tid, bool pf_next) {
  for (int i = tid; i < Dn * 32; i += 256) {
    int r = i >> 5, c = (i & 31) * 4;
    const float4 v = *(const float4*)(src + (size_t)r * Dn + c);
    bf16_t* d = dst + r * LS + c;
    d[0] = (bf16_t)v.x; d[1] = (bf16_t)v.y; d[2] = (bf16_t)v.z; d[3] = (bf16_t)v.w;
    if (pf_next) __builtin_prefetch(src + (size_t)(r + 128) * Dn + c, 0, 3);
  }
}

// ---- setup: fused weights E1=F1k@Wk, E2=F2k@Wk, G1=F1q@Wq, G2=F2q@Wq, bf16 Wv, Wo ----
__global__ __launch_bounds__(256) void k_wgen(
    const float* __restrict__ Wq, const float* __restrict__ Wk, const float* __restrict__ Wv,
    const float* __restrict__ fq1, const float* __restrict__ fq2,
    const float* __restrict__ fk1, const float* __restrict__ fk2,
    const float* __restrict__ Wo, bf16_t* __restrict__ dst) {
  const int mat = blockIdx.y;
  const int idx = blockIdx.x * 256 + threadIdx.x;  // 0..16383
  const int i = idx >> 7, j = idx & 127;
  float v = 0.f;
  if (mat < 4) {
    const float* F = (mat == 0) ? fk1 : (mat == 1) ? fk2 : (mat == 2) ? fq1 : fq2;
    const float* W = (mat < 2) ? Wk : Wq;
    for (int m = 0; m < Dn; ++m) v += F[i * Dn + m] * W[m * Dn + j];
  } else if (mat == 4) {
    v = Wv[idx];
  } else {
    v = Wo[idx];
  }
  dst[mat * Dn * Dn + idx] = (bf16_t)v;
}

__global__ void k_zero(float* __restrict__ p, int n) {
  int i = blockIdx.x * 256 + threadIdx.x;
  if (i < n) p[i] = 0.f;
}

// ---- phase 1: S[b] = phi_k(x)^T @ v(x), accumulated in regs, atomic-added as S^T[e][d] ----
__global__ __launch_bounds__(256) void k_phase1(
    const float* __restrict__ x, const bf16_t* __restrict__ wE1,
    const bf16_t* __restrict__ wE2, const bf16_t* __restrict__ wWv,
    const float* __restrict__ kb1, const float* __restrict__ kb2,
    float* __restrict__ ST) {
  __shared__ bf16_t sE1[Dn * LS], sE2[Dn * LS], sWv[Dn * LS];
  __shared__ bf16_t sX[Dn * LS], sPhiT[Dn * LS], sVT[Dn * LS];
  const int tid = threadIdx.x, wave = tid >> 5, lane = tid & 31;
  const int l16 = lane & 15, hi = lane >> 4;
  const int b = blockIdx.y, wg = blockIdx.x;

  ld_weight(sE1, wE1, tid);
  ld_weight(sE2, wE2, tid);
  ld_weight(sWv, wWv, tid);

  v8f Sacc[8];
#pragma unroll
  for (int n = 0; n < 8; ++n) Sacc[n] = splat8(0.f);

  const float* xb = x + (size_t)b * Tn * Dn;
  const int t0base = wg * (Tn / WGPB);

  for (int tt = 0; tt < TILES; ++tt) {
    const int t0 = t0base + tt * 128;
    __syncthreads();
    ld_f32_tile(sX, xb + (size_t)t0 * Dn, tid, tt + 1 < TILES);
    __syncthreads();

    v16bf ax[4];
#pragma unroll
    for (int kc = 0; kc < 4; ++kc) ax[kc] = load_a(sX, 16 * wave + l16, kc, hi);

#pragma unroll
    for (int nt = 0; nt < 8; ++nt) {
      const int dcol = nt * 16 + l16;
      v8f a1 = splat8(kb1[dcol]);
      v8f a2 = splat8(kb2[dcol]);
      v8f av = splat8(0.f);
#pragma unroll
      for (int kc = 0; kc < 4; ++kc) {
        a1 = wmma_bf16(ax[kc], load_b(sE1, dcol, kc, hi), a1);
        a2 = wmma_bf16(ax[kc], load_b(sE2, dcol, kc, hi), a2);
        av = wmma_bf16(ax[kc], load_b(sWv, dcol, kc, hi), av);
      }
      v8f phi = a1 * a2;
      // D-layout is a row of the transposed tile: packed store
      store8(sPhiT + dcol * LS + 16 * wave + 8 * hi, phi);
      store8(sVT + dcol * LS + 16 * wave + 8 * hi, av);
    }
    __syncthreads();

    // S[d][e] += sum_t phiT[d][t] * vT[e][t]   (A rows = d strip, B cols = e, K = t)
    v16bf ap[4];
#pragma unroll
    for (int kc = 0; kc < 4; ++kc) ap[kc] = load_a(sPhiT, 16 * wave + l16, kc, hi);
#pragma unroll
    for (int nt = 0; nt < 8; ++nt) {
      const int ecol = nt * 16 + l16;
#pragma unroll
      for (int kc = 0; kc < 4; ++kc)
        Sacc[nt] = wmma_bf16(ap[kc], load_b(sVT, ecol, kc, hi), Sacc[nt]);
    }
  }

  float* STb = ST + (size_t)b * Dn * Dn;  // stored as S^T[e][d]
  const int dbase = 16 * wave + 8 * hi;
#pragma unroll
  for (int nt = 0; nt < 8; ++nt) {
    const int ecol = nt * 16 + l16;
#pragma unroll
    for (int r = 0; r < 8; ++r) atomicAdd(STb + ecol * Dn + dbase + r, Sacc[nt][r]);
  }
}

// ---- phase 2: o = phi_q(x) @ S, RMSNorm, @ Wo^T, running max over T ----
__global__ __launch_bounds__(256) void k_phase2(
    const float* __restrict__ x, const bf16_t* __restrict__ wG1,
    const bf16_t* __restrict__ wG2, const bf16_t* __restrict__ wWo,
    const float* __restrict__ qb1, const float* __restrict__ qb2,
    const float* __restrict__ rmsw, const float* __restrict__ ST,
    float* __restrict__ pmax) {
  __shared__ bf16_t sG1[Dn * LS], sG2[Dn * LS], sWo[Dn * LS];
  __shared__ bf16_t sX[Dn * LS], sST[Dn * LS], sPhi[Dn * LS], sOn[Dn * LS];
  __shared__ float wmaxs[8 * Dn];
  const int tid = threadIdx.x, wave = tid >> 5, lane = tid & 31;
  const int l16 = lane & 15, hi = lane >> 4;
  const int b = blockIdx.y, wg = blockIdx.x;

  ld_weight(sG1, wG1, tid);
  ld_weight(sG2, wG2, tid);
  ld_weight(sWo, wWo, tid);
  ld_f32_tile(sST, ST + (size_t)b * Dn * Dn, tid, false);  // S^T[e][d] -> bf16 LDS

  v8f mx[8];
#pragma unroll
  for (int n = 0; n < 8; ++n) mx[n] = splat8(-3.402823466e38f);

  const float* xb = x + (size_t)b * Tn * Dn;
  const int t0base = wg * (Tn / WGPB);

  for (int tt = 0; tt < TILES; ++tt) {
    const int t0 = t0base + tt * 128;
    __syncthreads();
    ld_f32_tile(sX, xb + (size_t)t0 * Dn, tid, tt + 1 < TILES);
    __syncthreads();

    v16bf ax[4];
#pragma unroll
    for (int kc = 0; kc < 4; ++kc) ax[kc] = load_a(sX, 16 * wave + l16, kc, hi);

#pragma unroll
    for (int nt = 0; nt < 8; ++nt) {
      const int dcol = nt * 16 + l16;
      v8f a1 = splat8(qb1[dcol]);
      v8f a2 = splat8(qb2[dcol]);
#pragma unroll
      for (int kc = 0; kc < 4; ++kc) {
        a1 = wmma_bf16(ax[kc], load_b(sG1, dcol, kc, hi), a1);
        a2 = wmma_bf16(ax[kc], load_b(sG2, dcol, kc, hi), a2);
      }
      v8f phi = a1 * a2;
#pragma unroll
      for (int r = 0; r < 8; ++r)  // row-major scatter (wave-private rows)
        sPhi[(16 * wave + 8 * hi + r) * LS + dcol] = (bf16_t)phi[r];
    }
    __syncthreads();

    // o[t][e] = sum_d phi_q[t][d] * S[d][e]; B lane reads S^T row e (contiguous d)
    v16bf aq[4];
#pragma unroll
    for (int kc = 0; kc < 4; ++kc) aq[kc] = load_a(sPhi, 16 * wave + l16, kc, hi);
    v8f oacc[8];
#pragma unroll
    for (int nt = 0; nt < 8; ++nt) {
      const int ecol = nt * 16 + l16;
      oacc[nt] = splat8(0.f);
#pragma unroll
      for (int kc = 0; kc < 4; ++kc)
        oacc[nt] = wmma_bf16(aq[kc], load_b(sST, ecol, kc, hi), oacc[nt]);
    }

    // RMSNorm across e: per-component cross-lane sums within 16-lane groups
    v8f ss = splat8(0.f);
#pragma unroll
    for (int nt = 0; nt < 8; ++nt) ss += oacc[nt] * oacc[nt];
    float rstd[8];
#pragma unroll
    for (int r = 0; r < 8; ++r) {
      float s = ss[r];
      s += __shfl_xor(s, 1);
      s += __shfl_xor(s, 2);
      s += __shfl_xor(s, 4);
      s += __shfl_xor(s, 8);
      rstd[r] = rsqrtf(s * (1.0f / (float)Dn) + 1e-5f);
    }
#pragma unroll
    for (int nt = 0; nt < 8; ++nt) {
      const int ecol = nt * 16 + l16;
      const float g = rmsw[ecol];
#pragma unroll
      for (int r = 0; r < 8; ++r)
        sOn[(16 * wave + 8 * hi + r) * LS + ecol] = (bf16_t)(oacc[nt][r] * rstd[r] * g);
    }
    __syncthreads();

    // y = o_n @ Wo^T, fold into running column max
    v16bf ao[4];
#pragma unroll
    for (int kc = 0; kc < 4; ++kc) ao[kc] = load_a(sOn, 16 * wave + l16, kc, hi);
#pragma unroll
    for (int nt = 0; nt < 8; ++nt) {
      const int ncol = nt * 16 + l16;
      v8f y = splat8(0.f);
#pragma unroll
      for (int kc = 0; kc < 4; ++kc)
        y = wmma_bf16(ao[kc], load_b(sWo, ncol, kc, hi), y);
#pragma unroll
      for (int r = 0; r < 8; ++r) mx[nt][r] = fmaxf(mx[nt][r], y[r]);
    }
  }

  // reduce per-wave maxima -> per-WG partial max over this T chunk
#pragma unroll
  for (int nt = 0; nt < 8; ++nt) {
    float m = mx[nt][0];
#pragma unroll
    for (int r = 1; r < 8; ++r) m = fmaxf(m, mx[nt][r]);
    m = fmaxf(m, __shfl_xor(m, 16));
    if (hi == 0) wmaxs[wave * Dn + nt * 16 + l16] = m;
  }
  __syncthreads();
  if (tid < Dn) {
    float m = wmaxs[tid];
#pragma unroll
    for (int w = 1; w < 8; ++w) m = fmaxf(m, wmaxs[w * Dn + tid]);
    pmax[((size_t)b * WGPB + wg) * Dn + tid] = m;
  }
}

// ---- final: max over partials, out = pooled @ Wp^T + bp ----
__global__ __launch_bounds__(256) void k_final(
    const float* __restrict__ pmax, const float* __restrict__ Wp,
    const float* __restrict__ bp, float* __restrict__ out) {
  __shared__ float pooled[Dn];
  const int b = blockIdx.x, tid = threadIdx.x;
  if (tid < Dn) {
    float m = -3.402823466e38f;
    for (int w = 0; w < WGPB; ++w) m = fmaxf(m, pmax[((size_t)b * WGPB + w) * Dn + tid]);
    pooled[tid] = m;
  }
  __syncthreads();
  float acc = bp[tid];
  for (int d = 0; d < Dn; ++d) acc += pooled[d] * Wp[tid * Dn + d];
  out[(size_t)b * Hn + tid] = acc;
}

extern "C" void kernel_launch(void* const* d_in, const int* in_sizes, int n_in,
                              void* d_out, int out_size, void* d_ws, size_t ws_size,
                              hipStream_t stream) {
  (void)in_sizes; (void)n_in; (void)out_size; (void)ws_size;
  const float* x    = (const float*)d_in[0];
  const float* Wq   = (const float*)d_in[1];
  const float* Wk   = (const float*)d_in[2];
  const float* Wv   = (const float*)d_in[3];
  const float* fq1  = (const float*)d_in[4];
  const float* qb1  = (const float*)d_in[5];
  const float* fq2  = (const float*)d_in[6];
  const float* qb2  = (const float*)d_in[7];
  const float* fk1  = (const float*)d_in[8];
  const float* kb1  = (const float*)d_in[9];
  const float* fk2  = (const float*)d_in[10];
  const float* kb2  = (const float*)d_in[11];
  const float* rmsw = (const float*)d_in[12];
  const float* Wo   = (const float*)d_in[13];
  const float* Wp   = (const float*)d_in[14];
  const float* bp   = (const float*)d_in[15];
  float* out = (float*)d_out;

  char* ws = (char*)d_ws;
  bf16_t* wsW = (bf16_t*)ws;                                    // 6 * 128*128 bf16 = 196608 B
  float* ST   = (float*)(ws + 6 * Dn * Dn * sizeof(bf16_t));    // 16 * 128*128 f32 = 1 MiB (S^T per batch)
  float* pmax = (float*)(ws + 6 * Dn * Dn * sizeof(bf16_t)
                            + (size_t)Bn * Dn * Dn * sizeof(float));  // 16*32*128 f32

  k_wgen<<<dim3(64, 6), 256, 0, stream>>>(Wq, Wk, Wv, fq1, fq2, fk1, fk2, Wo, wsW);
  k_zero<<<(Bn * Dn * Dn) / 256, 256, 0, stream>>>(ST, Bn * Dn * Dn);
  k_phase1<<<dim3(WGPB, Bn), 256, 0, stream>>>(x, wsW, wsW + Dn * Dn, wsW + 4 * Dn * Dn,
                                               kb1, kb2, ST);
  k_phase2<<<dim3(WGPB, Bn), 256, 0, stream>>>(x, wsW + 2 * Dn * Dn, wsW + 3 * Dn * Dn,
                                               wsW + 5 * Dn * Dn, qb1, qb2, rmsw, ST, pmax);
  k_final<<<Bn, 256, 0, stream>>>(pmax, Wp, bp, out);
}
